// NVMixtralSparseMoeBlock_60292750901619
// MI455X (gfx1250) — matrix operand order repaired and checked by reading, not compiled, not run on hardware
//
#include <hip/hip_runtime.h>

// ---------------------------------------------------------------------------
// Mixtral sparse-MoE block for MI455X (gfx1250, wave32, WMMA + TDM + async LDS).
// Compute-bound (~361 GFLOP dense vs ~352 MB weight traffic -> ~1 kFLOP/B):
//  - matrix math:   v_wmma_f32_16x16x32_bf16 (f32 accum)
//  - weight tiles:  TENSOR_LOAD_TO_LDS (TDM, double-buffered, TENSORcnt)
//  - activations:   GLOBAL_LOAD_ASYNC_TO_LDS_B128 (double-buffered, ASYNCcnt)
// ---------------------------------------------------------------------------

typedef __bf16 bhalf;
typedef bhalf    v16bf    __attribute__((ext_vector_type(16)));
typedef float    v8f      __attribute__((ext_vector_type(8)));
typedef unsigned uint32x4 __attribute__((ext_vector_type(4)));
typedef int      int32x8  __attribute__((ext_vector_type(8)));
typedef int      int32x4  __attribute__((ext_vector_type(4)));

constexpr int Hdim = 1024;
constexpr int Fdim = 3584;
constexpr int Edim = 8;
constexpr int Ntok = 2048;   // B*S

constexpr int TM  = 64;      // token tile per block
constexpr int TN  = 64;      // output-col tile per block
constexpr int TK  = 32;      // K step (WMMA bf16 K)
constexpr int PAD = 40;      // bf16 LDS row pitch (80B: 16B-aligned rows)

// Flat LDS pointers carry the LDS byte offset in their low 32 bits
// (ISA 10.2, LDS aperture).
__device__ __forceinline__ unsigned lds_addr_of(const void* p) {
    return (unsigned)(unsigned long long)(uintptr_t)p;
}

// Per-lane async 16B copy global -> LDS (GVS mode: saddr base + 32b vgpr offset).
__device__ __forceinline__ void async_copy_b128(unsigned ldsDst, const void* base, int byteOff) {
    asm volatile("global_load_async_to_lds_b128 %0, %1, %2"
                 :: "v"(ldsDst), "v"(byteOff), "s"(base) : "memory");
}

// TDM: DMA a 32x64 fp32 tile (row stride = strideElems) into LDS, compact rows.
// D# per CDNA5 ISA ch.8: group0 = {count=1, lds_addr, global_addr, type=2},
// group1 = {data_size=4B, tensor_dim0=strideElems, tensor_dim1=32,
//           tile_dim0=64, tile_dim1=32, tensor_dim0_stride=strideElems}.
__device__ __forceinline__ void tdm_load_f32_tile(const float* tile, unsigned ldsAddr,
                                                  unsigned strideElems) {
    const unsigned long long ga = (unsigned long long)(uintptr_t)tile;
    uint32x4 g0;
    g0[0] = 1u;                                                 // count=1
    g0[1] = ldsAddr;                                            // lds_addr
    g0[2] = (unsigned)ga;                                       // global_addr[31:0]
    g0[3] = (unsigned)((ga >> 32) & 0x01FFFFFFu) | 0x80000000u; // addr[56:32] | type=2
    int32x8 g1;
    g1[0] = (int)(2u << 16);                                    // data_size=4B; no pad/iterate
    g1[1] = (int)((strideElems & 0xFFFFu) << 16);               // tensor_dim0 lo16
    g1[2] = (int)((strideElems >> 16) | (32u << 16));           // tensor_dim0 hi16 | tensor_dim1 lo16
    g1[3] = (int)(64u << 16);                                   // tensor_dim1 hi16=0 | tile_dim0=64
    g1[4] = 32;                                                 // tile_dim1=32, tile_dim2=0
    g1[5] = (int)strideElems;                                   // tensor_dim0_stride lo32
    g1[6] = 0;                                                  // stride hi16, dim1_stride lo16
    g1[7] = 0;
    const int32x4 z4 = {0, 0, 0, 0};
    const int32x8 z8 = {0, 0, 0, 0, 0, 0, 0, 0};
    __builtin_amdgcn_tensor_load_to_lds(g0, g1, z4, z4, z8, 0);
}

__device__ __forceinline__ unsigned pack_bf16(float a, float b) {
    union { bhalf h[2]; unsigned u; } p;
    p.h[0] = (bhalf)a; p.h[1] = (bhalf)b;
    return p.u;
}

// ---- WMMA fragment loaders (layouts per CDNA5 ISA 7.12.2, wave32) ---------
__device__ __forceinline__ v16bf load_a_frag(const bhalf* sA, int mBase, int lane) {
    const int m    = mBase + (lane & 15);
    const int half = lane >> 4;
    union { v16bf v; unsigned u[8]; } f;
    const unsigned* row = (const unsigned*)(sA + m * PAD);
#pragma unroll
    for (int r = 0; r < 8; ++r) {
        const int kPair = ((r < 4) ? 0 : 16) + half * 8 + (r & 3) * 2;
        f.u[r] = row[kPair >> 1];
    }
    return f.v;
}

__device__ __forceinline__ v16bf load_b_frag(const bhalf* sB, int nBase, int lane) {
    const int n    = nBase + (lane & 15);
    const int half = lane >> 4;
    union { v16bf v; unsigned u[8]; } f;
    const unsigned* row = (const unsigned*)(sB + n * PAD);
#pragma unroll
    for (int r = 0; r < 8; ++r) f.u[r] = row[(((r < 4) ? 0 : 16) + half * 8 + (r & 3) * 2) >> 1];
    return f.v;
}

// ---- Router: logits -> softmax -> top-2 -> renormalized combine[N,E] ------
__global__ __launch_bounds__(256) void moe_router(const float* __restrict__ x,
                                                  const float* __restrict__ gw,
                                                  float* __restrict__ combine) {
    const int wave = threadIdx.x >> 5;
    const int lane = threadIdx.x & 31;
    const int n    = blockIdx.x * 8 + wave;

    float acc[Edim] = {};
    for (int h = lane; h < Hdim; h += 32) {
        const float xv = x[(size_t)n * Hdim + h];
#pragma unroll
        for (int e = 0; e < Edim; ++e) acc[e] += xv * gw[h * Edim + e];
    }
#pragma unroll
    for (int e = 0; e < Edim; ++e) {
#pragma unroll
        for (int off = 16; off > 0; off >>= 1) acc[e] += __shfl_xor(acc[e], off, 32);
    }
    if (lane == 0) {
        int i1 = 0; float l1 = acc[0];
        for (int e = 1; e < Edim; ++e) if (acc[e] > l1) { l1 = acc[e]; i1 = e; }
        int i2 = -1; float l2 = -3.0e38f;
        for (int e = 0; e < Edim; ++e) if (e != i1 && acc[e] > l2) { l2 = acc[e]; i2 = e; }
        const float b  = __expf(l2 - l1);
        const float wA = 1.0f / (1.0f + b);
        const float wB = b / (1.0f + b);
#pragma unroll
        for (int e = 0; e < Edim; ++e)
            combine[(size_t)n * Edim + e] = (e == i1) ? wA : (e == i2 ? wB : 0.0f);
    }
}

// ---- fp32 -> bf16 activation convert (once) -------------------------------
__global__ __launch_bounds__(256) void cvt_bf16(const float* __restrict__ x,
                                                bhalf* __restrict__ xb, int n) {
    const int i = blockIdx.x * 256 + threadIdx.x;
    if (i < n) xb[i] = (bhalf)x[i];
}

// ---- GEMM1: hdn[N,F] = silu(x@w1e) * (x@w3e), bf16 out --------------------
__global__ __launch_bounds__(128) void moe_gemm1(const bhalf* __restrict__ xb,
                                                 const float* __restrict__ w1e,
                                                 const float* __restrict__ w3e,
                                                 bhalf* __restrict__ hdn) {
    __shared__ bhalf sA[2][TM * PAD];     // activations, bf16, double-buffered
    __shared__ float rB1[2][TK * TN];     // raw fp32 weight tiles (TDM targets)
    __shared__ float rB3[2][TK * TN];
    __shared__ bhalf sB1[TN * PAD];       // repacked bf16 [n][k]
    __shared__ bhalf sB3[TN * PAD];

    const int n0   = blockIdx.x * TN;
    const int m0   = blockIdx.y * TM;
    const int tid  = threadIdx.x;
    const int lane = tid & 31;
    const int wave = tid >> 5;

    v8f acc1[4] = {}; v8f acc3[4] = {};

    auto issueA = [&](int buf, int k0) {
        const unsigned sAoff = lds_addr_of(&sA[buf][0]);
        int idx = tid;
#pragma unroll
        for (int it = 0; it < 2; ++it, idx += 128) {
            const int row = idx >> 2, q = idx & 3;
            async_copy_b128(sAoff + (unsigned)(row * (PAD * 2) + q * 16),
                            xb, ((m0 + row) * Hdim + k0 + q * 8) * 2);
        }
    };
    auto issueB = [&](int buf, int k0) {
        tdm_load_f32_tile(w1e + (size_t)k0 * Fdim + n0, lds_addr_of(&rB1[buf][0]), Fdim);
        tdm_load_f32_tile(w3e + (size_t)k0 * Fdim + n0, lds_addr_of(&rB3[buf][0]), Fdim);
    };

    issueA(0, 0);
    if (wave == 0) issueB(0, 0);

    for (int k0 = 0, step = 0; k0 < Hdim; k0 += TK, ++step) {
        const int cur = step & 1, nxt = cur ^ 1;
        if (k0 + TK < Hdim) {
            issueA(nxt, k0 + TK);
            if (wave == 0) issueB(nxt, k0 + TK);
            asm volatile("s_wait_asynccnt 0x2" ::: "memory");    // oldest A buffer done
            if (wave == 0) __builtin_amdgcn_s_wait_tensorcnt(2); // oldest 2 TDM ops done
        } else {
            asm volatile("s_wait_asynccnt 0x0" ::: "memory");
            if (wave == 0) __builtin_amdgcn_s_wait_tensorcnt(0);
        }
        __syncthreads();

        // Repack raw fp32 tiles -> packed bf16, [n][k] layout for B fragments.
        {
            const int nn = tid & 63, p0 = tid >> 6;
#pragma unroll
            for (int p = p0; p < 16; p += 2) {
                const int kk = 2 * p;
                *(unsigned*)&sB1[nn * PAD + kk] =
                    pack_bf16(rB1[cur][kk * TN + nn], rB1[cur][(kk + 1) * TN + nn]);
                *(unsigned*)&sB3[nn * PAD + kk] =
                    pack_bf16(rB3[cur][kk * TN + nn], rB3[cur][(kk + 1) * TN + nn]);
            }
        }
        __syncthreads();

        const v16bf af = load_a_frag(&sA[cur][0], wave * 16, lane);
#pragma unroll
        for (int sub = 0; sub < 4; ++sub) {
            const v16bf b1 = load_b_frag(sB1, sub * 16, lane);
            acc1[sub] = __builtin_amdgcn_wmma_f32_16x16x32_bf16(
                false, af, false, b1, (short)0, acc1[sub], false, false);
            const v16bf b3 = load_b_frag(sB3, sub * 16, lane);
            acc3[sub] = __builtin_amdgcn_wmma_f32_16x16x32_bf16(
                false, af, false, b3, (short)0, acc3[sub], false, false);
        }
        __syncthreads();
    }

    // Epilogue: fused SiLU * gate, store bf16. C layout: M = half*8 + r, N = lane&15.
    const int half = lane >> 4, nl = lane & 15;
#pragma unroll
    for (int sub = 0; sub < 4; ++sub) {
#pragma unroll
        for (int r = 0; r < 8; ++r) {
            const float g = acc1[sub][r];
            const float u = acc3[sub][r];
            const float h = (g / (1.0f + __expf(-g))) * u;
            const int row = m0 + wave * 16 + half * 8 + r;
            const int col = n0 + sub * 16 + nl;
            hdn[(size_t)row * Fdim + col] = (bhalf)h;
        }
    }
}

// ---- GEMM2: out[N,H] += combine[:,e] * (hdn @ w2e) ------------------------
__global__ __launch_bounds__(128) void moe_gemm2(const bhalf* __restrict__ hdn,
                                                 const float* __restrict__ w2e,
                                                 const float* __restrict__ combine,
                                                 int expert,
                                                 float* __restrict__ out) {
    __shared__ bhalf sA[2][TM * PAD];
    __shared__ float rB[2][TK * TN];
    __shared__ bhalf sB[TN * PAD];

    const int n0   = blockIdx.x * TN;   // H cols
    const int m0   = blockIdx.y * TM;   // tokens
    const int tid  = threadIdx.x;
    const int lane = tid & 31;
    const int wave = tid >> 5;

    v8f acc[4] = {};

    auto issueA = [&](int buf, int k0) {
        const unsigned sAoff = lds_addr_of(&sA[buf][0]);
        int idx = tid;
#pragma unroll
        for (int it = 0; it < 2; ++it, idx += 128) {
            const int row = idx >> 2, q = idx & 3;
            async_copy_b128(sAoff + (unsigned)(row * (PAD * 2) + q * 16),
                            hdn, ((m0 + row) * Fdim + k0 + q * 8) * 2);
        }
    };

    issueA(0, 0);
    if (wave == 0)
        tdm_load_f32_tile(w2e + (size_t)0 * Hdim + n0, lds_addr_of(&rB[0][0]), Hdim);

    for (int k0 = 0, step = 0; k0 < Fdim; k0 += TK, ++step) {
        const int cur = step & 1, nxt = cur ^ 1;
        if (k0 + TK < Fdim) {
            issueA(nxt, k0 + TK);
            if (wave == 0)
                tdm_load_f32_tile(w2e + (size_t)(k0 + TK) * Hdim + n0,
                                  lds_addr_of(&rB[nxt][0]), Hdim);
            asm volatile("s_wait_asynccnt 0x2" ::: "memory");
            if (wave == 0) __builtin_amdgcn_s_wait_tensorcnt(1);
        } else {
            asm volatile("s_wait_asynccnt 0x0" ::: "memory");
            if (wave == 0) __builtin_amdgcn_s_wait_tensorcnt(0);
        }
        __syncthreads();

        {
            const int nn = tid & 63, p0 = tid >> 6;
#pragma unroll
            for (int p = p0; p < 16; p += 2) {
                const int kk = 2 * p;
                *(unsigned*)&sB[nn * PAD + kk] =
                    pack_bf16(rB[cur][kk * TN + nn], rB[cur][(kk + 1) * TN + nn]);
            }
        }
        __syncthreads();

        const v16bf af = load_a_frag(&sA[cur][0], wave * 16, lane);
#pragma unroll
        for (int sub = 0; sub < 4; ++sub) {
            const v16bf bf = load_b_frag(sB, sub * 16, lane);
            acc[sub] = __builtin_amdgcn_wmma_f32_16x16x32_bf16(
                false, af, false, bf, (short)0, acc[sub], false, false);
        }
        __syncthreads();
    }

    // Epilogue: scale rows by combine weight and accumulate into out.
    const int half = lane >> 4, nl = lane & 15;
    float cw[8];
#pragma unroll
    for (int r = 0; r < 8; ++r)
        cw[r] = combine[(size_t)(m0 + wave * 16 + half * 8 + r) * Edim + expert];
#pragma unroll
    for (int sub = 0; sub < 4; ++sub) {
#pragma unroll
        for (int r = 0; r < 8; ++r) {
            const int row = m0 + wave * 16 + half * 8 + r;
            const int col = n0 + sub * 16 + nl;
            const size_t o = (size_t)row * Hdim + col;
            out[o] += cw[r] * acc[sub][r];   // disjoint per block; experts stream-serialized
        }
    }
}

// ---------------------------------------------------------------------------
extern "C" void kernel_launch(void* const* d_in, const int* in_sizes, int n_in,
                              void* d_out, int out_size, void* d_ws, size_t ws_size,
                              hipStream_t stream) {
    const float* x  = (const float*)d_in[0];   // [2,1024,1024]
    const float* gw = (const float*)d_in[1];   // [1024,8]
    const float* w1 = (const float*)d_in[2];   // [8,1024,3584]
    const float* w3 = (const float*)d_in[3];   // [8,1024,3584]
    const float* w2 = (const float*)d_in[4];   // [8,3584,1024]
    float* out      = (float*)d_out;           // [2,1024,1024]

    char* ws = (char*)d_ws;
    bhalf* xb      = (bhalf*)(ws);                                          // 4 MB
    float* combine = (float*)(ws + (size_t)Ntok * Hdim * 2);                // 64 KB
    bhalf* hdn     = (bhalf*)(ws + (size_t)Ntok * Hdim * 2 + (size_t)Ntok * Edim * 4); // 14 MB

    (void)hipMemsetAsync(out, 0, (size_t)Ntok * Hdim * sizeof(float), stream);

    moe_router<<<Ntok / 8, 256, 0, stream>>>(x, gw, combine);
    cvt_bf16<<<(Ntok * Hdim) / 256, 256, 0, stream>>>(x, xb, Ntok * Hdim);

    for (int e = 0; e < Edim; ++e) {
        moe_gemm1<<<dim3(Fdim / TN, Ntok / TM), 128, 0, stream>>>(
            xb, w1 + (size_t)e * Hdim * Fdim, w3 + (size_t)e * Hdim * Fdim, hdn);
        moe_gemm2<<<dim3(Hdim / TN, Ntok / TM), 128, 0, stream>>>(
            hdn, w2 + (size_t)e * Fdim * Hdim, combine, e, out);
    }
}
